// GraphSAGE_8426725835327
// MI455X (gfx1250) — matrix operand order repaired and checked by reading, not compiled, not run on hardware
//
#include <hip/hip_runtime.h>
#include <hip/hip_bf16.h>

#define N_NODES 50000
#define N_EDGES 800000
#define DIM_IN  96
#define DIM_H   128
#define DIM_OUT 16

typedef __attribute__((ext_vector_type(16))) __bf16 v16bf;
typedef __attribute__((ext_vector_type(8)))  float  v8f;

// ---------------------------------------------------------------- utilities

__global__ void zero_f32(float* __restrict__ p, size_t n) {
    size_t i = (size_t)blockIdx.x * blockDim.x + threadIdx.x;
    size_t stride = (size_t)gridDim.x * blockDim.x;
    for (; i < n; i += stride) p[i] = 0.0f;
}

// Repack a row-major f32 weight [K, NC] into bf16 WMMA B-fragment order:
// out[((ks*numTiles + t)*32 + lane)*16 + e] = W[(ks*32 + (lane<16?0:16) + e)*NC + t*16 + (lane&15)]
// so each wave fragment is one contiguous 32-byte vector load per lane.
__global__ void pack_b_frags(const float* __restrict__ W, __bf16* __restrict__ out,
                             int K, int NC) {
    int idx = blockIdx.x * blockDim.x + threadIdx.x;
    if (idx >= K * NC) return;
    int e    = idx & 15;
    int lane = (idx >> 4) & 31;
    int rem  = idx >> 9;                 // ks*numTiles + t
    int numT = NC / 16;
    int t    = rem % numT;
    int ks   = rem / numT;
    int n    = lane & 15;
    int k    = ks * 32 + ((lane < 16) ? 0 : 16) + e;
    out[idx] = (__bf16)W[(size_t)k * NC + t * 16 + n];
}

__global__ void count_deg(const int* __restrict__ dst, float* __restrict__ deg) {
    int e = blockIdx.x * blockDim.x + threadIdx.x;
    if (e < N_EDGES) atomicAdd(&deg[dst[e]], 1.0f);
}

__global__ void inv_deg(float* __restrict__ deg) {
    int i = blockIdx.x * blockDim.x + threadIdx.x;
    if (i < N_NODES) deg[i] = 1.0f / fmaxf(deg[i], 1.0f);
}

// Scatter-add of per-edge source features; one thread per float4 chunk.
template <int DIM>
__global__ void scatter_add(const float* __restrict__ feat,
                            const int* __restrict__ src,
                            const int* __restrict__ dst,
                            float* __restrict__ agg) {
    const int CH = DIM / 4;
    long long tid = (long long)blockIdx.x * blockDim.x + threadIdx.x;
    long long e = tid / CH;
    if (e >= N_EDGES) return;
    int c = (int)(tid % CH) * 4;
    int s = src[e];
    int d = dst[e];
    const float4 v = *(const float4*)(feat + (size_t)s * DIM + c);
    float* o = agg + (size_t)d * DIM + c;
    atomicAdd(o + 0, v.x);
    atomicAdd(o + 1, v.y);
    atomicAdd(o + 2, v.z);
    atomicAdd(o + 3, v.w);
}

// ---------------------------------------------------------------- WMMA GEMMs
//
// One wave per 16-row output tile.
// A fragment layout (16-bit A, 16x32, contiguous k, same order as pack_b_frags):
//   lanes 0-15 : M = lane,    K = 0..15
//   lanes 16-31: M = lane-16, K = 16..31
// C/D: VGPR r holds row M = r + (lane<16 ? 0 : 8), col N = lane&15.

// LDS A tile: stride 36 floats (16B-aligned rows; 36m mod 64 distinct for all
// 16 rows -> conflict-free fragment gathers).
__device__ __forceinline__ v16bf build_a_frag(const float As[16][36], int lane) {
    const int m   = lane & 15;
    const int kbA = (lane < 16) ? 0 : 16;
    v16bf a;
#pragma unroll
    for (int e2 = 0; e2 < 16; ++e2) {
        int i = e2 >> 1, j = e2 & 1;
        int k = kbA + 2 * i + j;
        a[e2] = (__bf16)As[m][k];
    }
    return a;
}

// Stage a 16x32 f32 tile: 4 float4 loads + b128 LDS stores per lane,
// coalesced; unconditional speculative prefetch of the next k-step.
template <int LD>
__device__ __forceinline__ void stage_a(float As[16][36],
                                        const float* __restrict__ srcF,
                                        int row0, int k0, int lane) {
#pragma unroll
    for (int j = 0; j < 4; ++j) {
        int idx4 = j * 32 + lane;
        int r    = idx4 >> 3;
        int c4   = (idx4 & 7) * 4;
        const float4* gp = (const float4*)(srcF + (size_t)(row0 + r) * LD + k0 + c4);
        float4 v = *gp;
        __builtin_prefetch(gp + 8, 0, 3);      // 32 floats ahead = next k-step
        *(float4*)&As[r][c4] = v;
    }
}

// Layer 1: h = relu( (aggX*invd) @ W1_l + x @ W1_r + b1 ).
// Phase 0 accumulates aggX@W1_l, rows are then scaled by invd (row scaling
// commutes with @W), phase 1 accumulates x@W1_r on top.
__global__ __launch_bounds__(32) void gemm_layer1(
    const float* __restrict__ x, const float* __restrict__ aggX,
    const float* __restrict__ invd,
    const v16bf* __restrict__ Wlf, const v16bf* __restrict__ Wrf,
    const float* __restrict__ bias, float* __restrict__ hout) {
    __shared__ float As[16][36];
    const int row0 = blockIdx.x * 16;
    const int lane = threadIdx.x;
    const int n    = lane & 15;
    const int mhi  = (lane < 16) ? 0 : 8;

    v8f c[8] = {};

    // phase 0: c = aggX @ W1_l
    for (int ks = 0; ks < DIM_IN / 32; ++ks) {
        stage_a<DIM_IN>(As, aggX, row0, ks * 32, lane);
        __syncthreads();
        v16bf a = build_a_frag(As, lane);
        __syncthreads();
#pragma unroll
        for (int t = 0; t < 8; ++t) {
            v16bf b = Wlf[(ks * 8 + t) * 32 + lane];
            c[t] = __builtin_amdgcn_wmma_f32_16x16x32_bf16(
                false, a, false, b, (short)0, c[t], false, false);
        }
    }

    // mean: scale accumulator rows by 1/deg
    float dsc[8];
#pragma unroll
    for (int r = 0; r < 8; ++r) dsc[r] = invd[row0 + mhi + r];
#pragma unroll
    for (int t = 0; t < 8; ++t)
#pragma unroll
        for (int r = 0; r < 8; ++r) c[t][r] *= dsc[r];

    // phase 1: c += x @ W1_r
    for (int ks = 0; ks < DIM_IN / 32; ++ks) {
        stage_a<DIM_IN>(As, x, row0, ks * 32, lane);
        __syncthreads();
        v16bf a = build_a_frag(As, lane);
        __syncthreads();
#pragma unroll
        for (int t = 0; t < 8; ++t) {
            v16bf b = Wrf[(ks * 8 + t) * 32 + lane];
            c[t] = __builtin_amdgcn_wmma_f32_16x16x32_bf16(
                false, a, false, b, (short)0, c[t], false, false);
        }
    }

    // epilogue: bias + ReLU
#pragma unroll
    for (int t = 0; t < 8; ++t) {
        const int col = t * 16 + n;
        const float bb = bias[col];
#pragma unroll
        for (int r = 0; r < 8; ++r) {
            float v = c[t][r] + bb;
            hout[(size_t)(row0 + mhi + r) * DIM_H + col] = v > 0.0f ? v : 0.0f;
        }
    }
}

// p = h @ W2_l and q = h @ W2_r in one pass (shared A fragments).
// lin_l is pushed AHEAD of the mean aggregation (segment_mean(h)@W ==
// segment_mean(h@W)), so layer-2 edges move 16 floats instead of 128.
__global__ __launch_bounds__(32) void gemm_pq(
    const float* __restrict__ h,
    const v16bf* __restrict__ Wlf, const v16bf* __restrict__ Wrf,
    float* __restrict__ p, float* __restrict__ q) {
    __shared__ float As[16][36];
    const int row0 = blockIdx.x * 16;
    const int lane = threadIdx.x;
    const int n    = lane & 15;
    const int mhi  = (lane < 16) ? 0 : 8;

    v8f cp = {}, cq = {};
    for (int ks = 0; ks < DIM_H / 32; ++ks) {
        stage_a<DIM_H>(As, h, row0, ks * 32, lane);
        __syncthreads();
        v16bf a = build_a_frag(As, lane);
        __syncthreads();
        v16bf bl = Wlf[ks * 32 + lane];
        v16bf br = Wrf[ks * 32 + lane];
        cp = __builtin_amdgcn_wmma_f32_16x16x32_bf16(
            false, a, false, bl, (short)0, cp, false, false);
        cq = __builtin_amdgcn_wmma_f32_16x16x32_bf16(
            false, a, false, br, (short)0, cq, false, false);
    }
#pragma unroll
    for (int r = 0; r < 8; ++r) {
        size_t off = (size_t)(row0 + mhi + r) * DIM_OUT + n;
        p[off] = cp[r];
        q[off] = cq[r];
    }
}

// out = softmax( q + aggP*invd + b2 )  -- pure elementwise + 16-wide reduce.
// 16 lanes per row; shuffle masks 1..8 stay within each 16-lane group.
__global__ void softmax_out(const float* __restrict__ q,
                            const float* __restrict__ aggP,
                            const float* __restrict__ invd,
                            const float* __restrict__ bias,
                            float* __restrict__ out) {
    int tid = blockIdx.x * blockDim.x + threadIdx.x;
    int row = tid >> 4;
    int n   = tid & 15;
    if (row >= N_NODES) return;
    size_t off = (size_t)row * DIM_OUT + n;
    float v = q[off] + aggP[off] * invd[row] + bias[n];
    float mx = v;
#pragma unroll
    for (int msk = 1; msk <= 8; msk <<= 1)
        mx = fmaxf(mx, __shfl_xor(mx, msk, 32));
    float ex = expf(v - mx);
    float sm = ex;
#pragma unroll
    for (int msk = 1; msk <= 8; msk <<= 1)
        sm += __shfl_xor(sm, msk, 32);
    out[off] = ex / sm;
}

// ---------------------------------------------------------------- launcher

extern "C" void kernel_launch(void* const* d_in, const int* in_sizes, int n_in,
                              void* d_out, int out_size, void* d_ws, size_t ws_size,
                              hipStream_t stream) {
    (void)in_sizes; (void)n_in; (void)out_size; (void)ws_size;

    const float* x   = (const float*)d_in[0];
    const int*   ei  = (const int*)d_in[1];   // [2, E] int32 (jax x64 off)
    const float* W1l = (const float*)d_in[2];
    const float* b1  = (const float*)d_in[3];
    const float* W1r = (const float*)d_in[4];
    const float* W2l = (const float*)d_in[5];
    const float* b2  = (const float*)d_in[6];
    const float* W2r = (const float*)d_in[7];

    const int* srcI = ei;
    const int* dstI = ei + N_EDGES;

    // ws layout (f32): deg | aggX[N,96] | aggP[N,16] | h[N,128] | p[N,16] |
    //                  q[N,16] | bf16 fragment-packed weights
    float* ws   = (float*)d_ws;
    float* deg  = ws;
    float* aggX = deg + N_NODES;
    float* aggP = aggX + (size_t)N_NODES * DIM_IN;
    float* h    = aggP + (size_t)N_NODES * DIM_OUT;
    float* p    = h + (size_t)N_NODES * DIM_H;
    float* q    = p + (size_t)N_NODES * DIM_OUT;
    __bf16* w1l = (__bf16*)(q + (size_t)N_NODES * DIM_OUT);  // 32B-aligned
    __bf16* w1r = w1l + DIM_IN * DIM_H;
    __bf16* w2l = w1r + DIM_IN * DIM_H;
    __bf16* w2r = w2l + DIM_H * DIM_OUT;

    // zero deg + aggX + aggP (h/p/q fully overwritten)
    size_t zn = (size_t)N_NODES * (1 + DIM_IN + DIM_OUT);
    zero_f32<<<2048, 256, 0, stream>>>(ws, zn);

    // fragment-packed bf16 weights
    pack_b_frags<<<(DIM_IN * DIM_H + 255) / 256, 256, 0, stream>>>(W1l, w1l, DIM_IN, DIM_H);
    pack_b_frags<<<(DIM_IN * DIM_H + 255) / 256, 256, 0, stream>>>(W1r, w1r, DIM_IN, DIM_H);
    pack_b_frags<<<(DIM_H * DIM_OUT + 255) / 256, 256, 0, stream>>>(W2l, w2l, DIM_H, DIM_OUT);
    pack_b_frags<<<(DIM_H * DIM_OUT + 255) / 256, 256, 0, stream>>>(W2r, w2r, DIM_H, DIM_OUT);

    // degree count and inverse
    count_deg<<<(N_EDGES + 255) / 256, 256, 0, stream>>>(dstI, deg);
    inv_deg<<<(N_NODES + 255) / 256, 256, 0, stream>>>(deg);

    // layer-1 aggregation: aggX = sum_j x[src]
    {
        long long tot = (long long)N_EDGES * (DIM_IN / 4);
        scatter_add<DIM_IN><<<(int)((tot + 255) / 256), 256, 0, stream>>>(x, srcI, dstI, aggX);
    }

    // layer-1 fused GEMM (+bias+ReLU)
    gemm_layer1<<<N_NODES / 16, 32, 0, stream>>>(x, aggX, deg,
                                                 (const v16bf*)w1l, (const v16bf*)w1r, b1, h);

    // p = h@W2_l, q = h@W2_r in one pass
    gemm_pq<<<N_NODES / 16, 32, 0, stream>>>(h, (const v16bf*)w2l, (const v16bf*)w2r, p, q);

    // layer-2 aggregation on 16-dim p (8x less traffic/atomics than on h)
    {
        long long tot = (long long)N_EDGES * (DIM_OUT / 4);
        scatter_add<DIM_OUT><<<(int)((tot + 255) / 256), 256, 0, stream>>>(p, srcI, dstI, aggP);
    }

    // out = softmax( q + aggP*invd + b2 )
    softmax_out<<<(N_NODES * DIM_OUT + 255) / 256, 256, 0, stream>>>(q, aggP, deg, b2, (float*)d_out);
}
